// GAEM_26920855011384
// MI455X (gfx1250) — matrix operand order repaired and checked by reading, not compiled, not run on hardware
//
#include <hip/hip_runtime.h>
#include <math.h>

typedef _Float16 half_t;
typedef __attribute__((ext_vector_type(16))) _Float16 v16h;
typedef __attribute__((ext_vector_type(8)))  _Float16 v8h;
typedef __attribute__((ext_vector_type(8)))  float    v8f;

// ---------------------------------------------------------------------------
// WMMA: D(16x16 f32) = A(16x32 f16) * B(32x16 f16) + C
// ---------------------------------------------------------------------------
static __device__ __forceinline__ v8f wmma_f16(v16h a, v16h b, v8f c) {
  return __builtin_amdgcn_wmma_f32_16x16x32_f16(
      false, a, false, b, (short)0, c, false, false);
}

// A fragment: lane holds row tM+lo; K chunks [k0+8*hi, +7] and [k0+16+8*hi, +7]
static __device__ __forceinline__ v16h ldA(const half_t* __restrict__ arow,
                                           int k0, int hi) {
  v8h a0 = *(const v8h*)(arow + k0 + (hi << 3));
  v8h a1 = *(const v8h*)(arow + k0 + 16 + (hi << 3));
  return __builtin_shufflevector(a0, a1, 0, 1, 2, 3, 4, 5, 6, 7,
                                 8, 9, 10, 11, 12, 13, 14, 15);
}

static __device__ __forceinline__ float lrelu(float v) {
  return v > 0.f ? v : 0.01f * v;
}

// ---------------------------------------------------------------------------
// Graph preparation
// ---------------------------------------------------------------------------
__global__ __launch_bounds__(256) void k_prep(const float* __restrict__ f,
                                              float* __restrict__ pp,
                                              int total, int C) {
  int i = blockIdx.x * blockDim.x + threadIdx.x;
  if (i >= total) return;
  const float* p = f + (long long)i * C;
  float th = p[0], ph = p[1], rad = p[2];
  float h = rad * sinf(ph);
  float a = h * tanf(th);
  float* o = pp + (long long)i * 8;
  o[0] = th; o[1] = h; o[2] = a; o[3] = p[3]; o[4] = p[4];
}

static __device__ __forceinline__ float dis_raw(const float* __restrict__ pp,
                                                long long pn, long long pm) {
  const float* A  = pp + pn * 8;
  const float* Bp = pp + pm * 8;
  float hn = A[1], hm = Bp[1];
  float x2y2 = hn * hn + hm * hm - hn * hm * cosf(A[0] - Bp[0]);
  float dz = A[2] - Bp[2];
  return x2y2 + dz * dz;
}

__global__ void k_mm_init(unsigned* mm) {
  mm[0] = 0x7F800000u;  // +inf
  mm[1] = 0u;           // 0.0f
}

__global__ __launch_bounds__(256) void k_mm(const float* __restrict__ pp,
                                            unsigned* __restrict__ mm,
                                            int B, int N) {
  long long total = (long long)B * N * N;
  long long stride = (long long)gridDim.x * blockDim.x;
  float lmn = INFINITY, lmx = 0.f;
  for (long long p = (long long)blockIdx.x * blockDim.x + threadIdx.x;
       p < total; p += stride) {
    long long b = p / ((long long)N * N);
    long long rem = p - b * (long long)N * N;
    long long n = rem / N, m = rem - n * (long long)N;
    float d = dis_raw(pp, b * N + n, b * N + m);
    lmn = fminf(lmn, d);
    lmx = fmaxf(lmx, d);
  }
  __shared__ float smn[256], smx[256];
  int t = threadIdx.x;
  smn[t] = lmn; smx[t] = lmx;
  __syncthreads();
  for (int off = 128; off > 0; off >>= 1) {
    if (t < off) {
      smn[t] = fminf(smn[t], smn[t + off]);
      smx[t] = fmaxf(smx[t], smx[t + off]);
    }
    __syncthreads();
  }
  if (t == 0) {
    atomicMin(&mm[0], __float_as_uint(smn[0]));
    atomicMax(&mm[1], __float_as_uint(smx[0]));
  }
}

// dis (1-normalized), par, and g0 = 0.5*(dis+par), all f16
__global__ __launch_bounds__(256) void k_graphs(const float* __restrict__ pp,
                                                const unsigned* __restrict__ mm,
                                                half_t* __restrict__ dis,
                                                half_t* __restrict__ par,
                                                half_t* __restrict__ g0,
                                                int B, int N) {
  long long p = (long long)blockIdx.x * blockDim.x + threadIdx.x;
  long long total = (long long)B * N * N;
  if (p >= total) return;
  long long b = p / ((long long)N * N);
  long long rem = p - b * (long long)N * N;
  long long n = rem / N, m = rem - n * (long long)N;
  long long pn = b * N + n, pm = b * N + m;
  float mn = __uint_as_float(mm[0]);
  float mx = __uint_as_float(mm[1]);
  float d = dis_raw(pp, pn, pm);
  float dv = 1.f - (d - mn) / (mx - mn);
  float c3n = pp[pn * 8 + 3], c3m = pp[pm * 8 + 3];
  float c4n = pp[pn * 8 + 4], c4m = pp[pm * 8 + 4];
  float mv3 = sqrtf(c3n * c3m);
  float mv4 = sqrtf(c4n * c4m);
  float p3 = ((mv3 - truncf(mv3)) == 0.f) ? 1.f : 0.f;
  float p4 = ((mv4 - truncf(mv4)) == 0.f) ? 1.f : 0.f;
  float pv = 0.5f * (p3 + p4);
  dis[p] = (half_t)dv;
  par[p] = (half_t)pv;
  g0[p]  = (half_t)(0.5f * (dv + pv));
}

// ---------------------------------------------------------------------------
// Guard-free f16 GEMM via WMMA. One wave computes a 16x64 output tile.
// A: f16 row-major (M x K), lda. Bt: f16 col-major (Ncols x K), ldb.
// Epilogue: f32 out (Cf) and/or f16 out (Ch, optionally transposed).
// grid = (M/16, Ncols/64, batch); K % 32 == 0.
// ---------------------------------------------------------------------------
__global__ __launch_bounds__(32) void k_gemm_h(
    const half_t* __restrict__ A, int lda, long long sA,
    const half_t* __restrict__ Bt, int ldb, long long sB,
    const float* __restrict__ bias,
    float* __restrict__ Cf, int ldc, long long sC,
    half_t* __restrict__ Ch, int ldh, long long sH, int transH,
    int K, float alpha, int act) {
  int lane = threadIdx.x & 31;
  int lo = lane & 15, hi = lane >> 4;
  int tM = blockIdx.x << 4;
  int tN = blockIdx.y << 6;
  int b = blockIdx.z;
  const half_t* arow = A + (long long)b * sA + (long long)(tM + lo) * lda;
  const half_t* b0   = Bt + (long long)b * sB + (long long)(tN + lo) * ldb;
  v8f acc[4] = {};
  for (int k0 = 0; k0 < K; k0 += 32) {
    __builtin_prefetch(arow + k0 + 128, 0, 0);
    v16h av = ldA(arow, k0, hi);
#pragma unroll
    for (int j = 0; j < 4; ++j) {
      v16h bv = ldA(b0 + (long long)(j << 4) * ldb, k0, hi);
      acc[j] = wmma_f16(av, bv, acc[j]);
    }
  }
#pragma unroll
  for (int j = 0; j < 4; ++j) {
    int cc = tN + (j << 4) + lo;
#pragma unroll
    for (int r = 0; r < 8; ++r) {
      int rr = tM + r + (hi << 3);
      float v = acc[j][r] * alpha;
      if (bias) v += bias[cc];
      if (act) v = lrelu(v);
      if (Cf) Cf[(long long)b * sC + (long long)rr * ldc + cc] = v;
      if (Ch) {
        long long o = transH ? ((long long)b * sH + (long long)cc * ldh + rr)
                             : ((long long)b * sH + (long long)rr * ldh + cc);
        Ch[o] = (half_t)v;
      }
    }
  }
}

// ---------------------------------------------------------------------------
// conv1d (SAME, kernel 16, D=256) as guard-free im2col WMMA GEMM.
// Xp: (B, N+16, 256) f16, rows 0..7 and N+8..N+15 are zeros.
// Wt: (256, 4096) f16 col-major packed: Wt[o*4096 + t*256 + ch] = w[o][ch][t].
// Y[b,n,o] = sum X[b,n-7+t,ch]*w[o,ch,t] + bias[o] (+ addT) -> f16 out.
// grid = (N/16, 4, B)
// ---------------------------------------------------------------------------
__global__ __launch_bounds__(32) void k_conv_h(
    const half_t* __restrict__ Xp, const half_t* __restrict__ Wt,
    const float* __restrict__ bias, const float* __restrict__ addT,
    half_t* __restrict__ Yh, int transH, int N) {
  int lane = threadIdx.x & 31;
  int lo = lane & 15, hi = lane >> 4;
  int tM = blockIdx.x << 4;
  int tN = blockIdx.y << 6;
  int b = blockIdx.z;
  int row = tM + lo;
  const half_t* Xb = Xp + (long long)b * (N + 16) * 256;
  const half_t* w0 = Wt + (long long)(tN + lo) * 4096;
  v8f acc[4] = {};
  for (int k0 = 0; k0 < 4096; k0 += 32) {
    int c0 = k0 + (hi << 3);
    int c1 = c0 + 16;
    v8h a0 = *(const v8h*)(Xb + (long long)(row + 1 + (c0 >> 8)) * 256 + (c0 & 255));
    v8h a1 = *(const v8h*)(Xb + (long long)(row + 1 + (c1 >> 8)) * 256 + (c1 & 255));
    v16h av = __builtin_shufflevector(a0, a1, 0, 1, 2, 3, 4, 5, 6, 7,
                                      8, 9, 10, 11, 12, 13, 14, 15);
    int kb = k0 + (hi << 3);
#pragma unroll
    for (int j = 0; j < 4; ++j) {
      const half_t* wj = w0 + (long long)(j << 4) * 4096 + kb;
      v16h bv = __builtin_shufflevector(*(const v8h*)wj, *(const v8h*)(wj + 16),
                                        0, 1, 2, 3, 4, 5, 6, 7,
                                        8, 9, 10, 11, 12, 13, 14, 15);
      acc[j] = wmma_f16(av, bv, acc[j]);
    }
  }
#pragma unroll
  for (int j = 0; j < 4; ++j) {
    int cc = tN + (j << 4) + lo;
#pragma unroll
    for (int r = 0; r < 8; ++r) {
      int rr = tM + r + (hi << 3);
      float v = acc[j][r] + bias[cc];
      if (addT) v += addT[((long long)b * N + rr) * 256 + cc];
      long long o = transH ? ((long long)b * 256 * N + (long long)cc * N + rr)
                           : (((long long)b * N + rr) * 256 + cc);
      Yh[o] = (half_t)v;
    }
  }
}

// ---------------------------------------------------------------------------
// Row softmax of (scores*dis + par) -> attn f16. One block per row.
// ---------------------------------------------------------------------------
__global__ __launch_bounds__(256) void k_softmax(
    const float* __restrict__ sc, const half_t* __restrict__ dis,
    const half_t* __restrict__ par, half_t* __restrict__ attn, int N) {
  long long row = blockIdx.x;
  const float* s = sc + row * N;
  const half_t* dr = dis + row * N;
  const half_t* pr = par + row * N;
  half_t* at = attn + row * N;
  __shared__ float sv[2048];
  __shared__ float red[256];
  int t = threadIdx.x;
  float lmax = -INFINITY;
  for (int m = t; m < N; m += 256) {
    float v = s[m] * (float)dr[m] + (float)pr[m];
    sv[m] = v;
    lmax = fmaxf(lmax, v);
  }
  red[t] = lmax;
  __syncthreads();
  for (int off = 128; off > 0; off >>= 1) {
    if (t < off) red[t] = fmaxf(red[t], red[t + off]);
    __syncthreads();
  }
  float mx = red[0];
  __syncthreads();
  float lsum = 0.f;
  for (int m = t; m < N; m += 256) {
    float e = __expf(sv[m] - mx);
    sv[m] = e;
    lsum += e;
  }
  red[t] = lsum;
  __syncthreads();
  for (int off = 128; off > 0; off >>= 1) {
    if (t < off) red[t] += red[t + off];
    __syncthreads();
  }
  float inv = 1.f / red[0];
  for (int m = t; m < N; m += 256) at[m] = (half_t)(sv[m] * inv);
}

// ---------------------------------------------------------------------------
// Out = LayerNorm(X (+R)) * g + b ; one wave per row (D = 256, wave32)
// ---------------------------------------------------------------------------
__global__ __launch_bounds__(32) void k_add_ln(
    const float* __restrict__ X, const float* __restrict__ R,
    const float* __restrict__ g, const float* __restrict__ be,
    float* __restrict__ Out, int D) {
  long long row = blockIdx.x;
  const float* x = X + row * D;
  const float* r = R ? R + row * D : nullptr;
  int lane = threadIdx.x;
  float vals[8];
  float s = 0.f, s2 = 0.f;
#pragma unroll
  for (int j = 0; j < 8; ++j) {
    int c = lane + j * 32;
    float v = x[c] + (r ? r[c] : 0.f);
    vals[j] = v;
    s += v;
    s2 += v * v;
  }
#pragma unroll
  for (int m = 16; m > 0; m >>= 1) {
    s += __shfl_xor(s, m);
    s2 += __shfl_xor(s2, m);
  }
  float mean = s / D;
  float var = s2 / D - mean * mean;
  float inv = rsqrtf(var + 1e-5f);
#pragma unroll
  for (int j = 0; j < 8; ++j) {
    int c = lane + j * 32;
    Out[row * D + c] = (vals[j] - mean) * inv * g[c] + be[c];
  }
}

// ---------------------------------------------------------------------------
// Pack kernels (f32 -> f16, padding / transpose)
// ---------------------------------------------------------------------------
// rows x Cs f32 -> rows x 32 f16, zero-padded columns
__global__ __launch_bounds__(256) void k_pack_pad32(const float* __restrict__ src,
                                                    half_t* __restrict__ dst,
                                                    long long rows, int Cs) {
  long long i = (long long)blockIdx.x * 256 + threadIdx.x;
  if (i >= rows * 32) return;
  long long r = i >> 5;
  int c = (int)(i & 31);
  dst[i] = (c < Cs) ? (half_t)src[r * Cs + c] : (half_t)0.f;
}

// w (K x Nc) f32 row-major -> wt (Nc x Kp) f16 col-major, K zero-padded to Kp
__global__ __launch_bounds__(256) void k_packw(const float* __restrict__ src,
                                               half_t* __restrict__ dst,
                                               int K, int Nc, int Kp) {
  int i = blockIdx.x * 256 + threadIdx.x;
  if (i >= Nc * Kp) return;
  int col = i / Kp, k = i - col * Kp;
  dst[i] = (k < K) ? (half_t)src[(long long)k * Nc + col] : (half_t)0.f;
}

// conv weight (256,256,16) -> (o, t*256+ch) f16 col-major
__global__ __launch_bounds__(256) void k_packcw(const float* __restrict__ src,
                                                half_t* __restrict__ dst) {
  int i = blockIdx.x * 256 + threadIdx.x;
  if (i >= 256 * 256 * 16) return;
  int o = i >> 12, rem = i & 4095, t = rem >> 8, ch = rem & 255;
  dst[i] = (half_t)src[(((long long)o * 256 + ch) << 4) + t];
}

// x (B,N,256) f32 -> Xp (B,N+16,256) f16 with 8 zero rows each side
__global__ __launch_bounds__(256) void k_packx(const float* __restrict__ X,
                                               half_t* __restrict__ Xp,
                                               int N, long long total) {
  long long i = (long long)blockIdx.x * 256 + threadIdx.x;
  if (i >= total) return;
  long long brow = i >> 8;
  int d = (int)(i & 255);
  long long b = brow / (N + 16);
  int rp = (int)(brow - b * (N + 16));
  float v = 0.f;
  if (rp >= 8 && rp < N + 8) v = X[((long long)b * N + (rp - 8)) * 256 + d];
  Xp[i] = (half_t)v;
}

__global__ __launch_bounds__(256) void k_pack(const float* __restrict__ src,
                                              half_t* __restrict__ dst,
                                              long long n) {
  long long i = (long long)blockIdx.x * 256 + threadIdx.x;
  if (i < n) dst[i] = (half_t)src[i];
}

// ---------------------------------------------------------------------------
// Host orchestration
// ---------------------------------------------------------------------------
extern "C" void kernel_launch(void* const* d_in, const int* in_sizes, int n_in,
                              void* d_out, int out_size, void* d_ws,
                              size_t ws_size, hipStream_t stream) {
  (void)in_sizes; (void)n_in; (void)out_size; (void)ws_size;
  const int B = 8, N = 2048, C = 8, D = 256, L = 3, KC = 16;

  const float* features = (const float*)d_in[0];
  const float* lst      = (const float*)d_in[1];
  const float* w_gcn    = (const float*)d_in[2];
  const float* lw1 = (const float*)d_in[3];  const float* lb1 = (const float*)d_in[4];
  const float* lw2 = (const float*)d_in[5];  const float* lb2 = (const float*)d_in[6];
  const float* lw3 = (const float*)d_in[7];  const float* lb3 = (const float*)d_in[8];
  const float* wq  = (const float*)d_in[9];  const float* bq  = (const float*)d_in[10];
  const float* wk  = (const float*)d_in[11]; const float* bk  = (const float*)d_in[12];
  const float* wv  = (const float*)d_in[13]; const float* bv  = (const float*)d_in[14];
  const float* mw1 = (const float*)d_in[15]; const float* mb1 = (const float*)d_in[16];
  const float* mw2 = (const float*)d_in[17]; const float* mb2 = (const float*)d_in[18];
  const float* mw3 = (const float*)d_in[19]; const float* mb3 = (const float*)d_in[20];
  const float* g1  = (const float*)d_in[21]; const float* b1  = (const float*)d_in[22];
  const float* g2  = (const float*)d_in[23]; const float* b2  = (const float*)d_in[24];
  const float* ow1 = (const float*)d_in[25]; const float* ob1 = (const float*)d_in[26];
  const float* og  = (const float*)d_in[27]; const float* ob  = (const float*)d_in[28];
  const float* ow2 = (const float*)d_in[29]; const float* ob2 = (const float*)d_in[30];

  char* ws = (char*)d_ws;
  size_t off = 0;
  auto alloc = [&](size_t bytes) -> char* {
    char* p = ws + off;
    off += (bytes + 255) & ~(size_t)255;
    return p;
  };
  const long long NN = (long long)N * N;
  const long long ND = (long long)N * D;
  const long long NP = (long long)(N + 16) * D;  // padded row stride (per batch)

  float*    pp     = (float*)alloc((size_t)B * N * 8 * 4);
  unsigned* mm     = (unsigned*)alloc(256);
  half_t*   dis    = (half_t*)alloc((size_t)B * NN * 2);
  half_t*   par    = (half_t*)alloc((size_t)B * NN * 2);
  half_t*   g0     = (half_t*)alloc((size_t)B * NN * 2);
  half_t*   attn   = (half_t*)alloc((size_t)B * NN * 2);
  float*    scores = (float*)alloc((size_t)B * NN * 4);
  float*    lstA   = (float*)alloc((size_t)B * ND * 4);
  float*    lstB   = (float*)alloc((size_t)B * ND * 4);
  float*    xbuf   = (float*)alloc((size_t)B * ND * 4);
  float*    t0     = (float*)alloc((size_t)B * ND * 4);
  // f16 activations
  half_t*   lsth   = (half_t*)alloc((size_t)B * N * 32 * 2);
  half_t*   feath  = (half_t*)alloc((size_t)B * N * 32 * 2);
  half_t*   l1h    = (half_t*)alloc((size_t)B * ND * 2);
  half_t*   l2h    = (half_t*)alloc((size_t)B * ND * 2);
  half_t*   xph    = (half_t*)alloc((size_t)B * NP * 2);
  half_t*   qh     = (half_t*)alloc((size_t)B * ND * 2);
  half_t*   kh     = (half_t*)alloc((size_t)B * ND * 2);
  half_t*   vth    = (half_t*)alloc((size_t)B * ND * 2);  // (B,D,N)
  half_t*   supT   = (half_t*)alloc((size_t)B * ND * 2);  // (B,D,N)
  half_t*   m1h    = (half_t*)alloc((size_t)B * ND * 2);
  half_t*   m2h    = (half_t*)alloc((size_t)B * ND * 2);
  half_t*   th0    = (half_t*)alloc((size_t)B * ND * 2);
  // f16 weights
  half_t*   lw1h   = (half_t*)alloc((size_t)D * 32 * 2);
  half_t*   lw2h   = (half_t*)alloc((size_t)D * D * 2);
  half_t*   lw3h   = (half_t*)alloc((size_t)D * D * 2);
  half_t*   wgcnh  = (half_t*)alloc((size_t)D * 32 * 2);
  half_t*   ow1h   = (half_t*)alloc((size_t)D * D * 2);
  half_t*   ow2h   = (half_t*)alloc((size_t)D * 256 * 2);
  half_t*   mwh[3][3];
  half_t*   cwh[3][3];  // [layer][q/k/v]
  for (int i = 0; i < 3; ++i)
    for (int j = 0; j < 3; ++j) {
      mwh[i][j] = (half_t*)alloc((size_t)D * D * 2);
      cwh[i][j] = (half_t*)alloc((size_t)D * D * KC * 2);
    }

  dim3 wv32(32);

  // --- graphs ---
  k_prep<<<(B * N + 255) / 256, 256, 0, stream>>>(features, pp, B * N, C);
  k_mm_init<<<1, 1, 0, stream>>>(mm);
  k_mm<<<4096, 256, 0, stream>>>(pp, mm, B, N);
  k_graphs<<<(unsigned)((B * NN + 255) / 256), 256, 0, stream>>>(pp, mm, dis, par, g0, B, N);

  // --- packs ---
  k_pack_pad32<<<(unsigned)(((long long)B * N * 32 + 255) / 256), 256, 0, stream>>>(lst, lsth, (long long)B * N, C + 1);
  k_pack_pad32<<<(unsigned)(((long long)B * N * 32 + 255) / 256), 256, 0, stream>>>(features, feath, (long long)B * N, C);
  k_packw<<<(D * 32 + 255) / 256, 256, 0, stream>>>(lw1, lw1h, C + 1, D, 32);
  k_packw<<<(D * D + 255) / 256, 256, 0, stream>>>(lw2, lw2h, D, D, D);
  k_packw<<<(D * D + 255) / 256, 256, 0, stream>>>(lw3, lw3h, D, D, D);
  k_packw<<<(D * 32 + 255) / 256, 256, 0, stream>>>(w_gcn, wgcnh, C, D, 32);
  k_packw<<<(D * D + 255) / 256, 256, 0, stream>>>(ow1, ow1h, D, D, D);
  k_packw<<<(D * 256 + 255) / 256, 256, 0, stream>>>(ow2, ow2h, D, 256, D);
  for (int i = 0; i < L; ++i) {
    k_packw<<<(D * D + 255) / 256, 256, 0, stream>>>(mw1 + (size_t)i * D * D, mwh[i][0], D, D, D);
    k_packw<<<(D * D + 255) / 256, 256, 0, stream>>>(mw2 + (size_t)i * D * D, mwh[i][1], D, D, D);
    k_packw<<<(D * D + 255) / 256, 256, 0, stream>>>(mw3 + (size_t)i * D * D, mwh[i][2], D, D, D);
    k_packcw<<<(D * D * KC + 255) / 256, 256, 0, stream>>>(wq + (size_t)i * D * D * KC, cwh[i][0]);
    k_packcw<<<(D * D * KC + 255) / 256, 256, 0, stream>>>(wk + (size_t)i * D * D * KC, cwh[i][1]);
    k_packcw<<<(D * D * KC + 255) / 256, 256, 0, stream>>>(wv + (size_t)i * D * D * KC, cwh[i][2]);
  }

  // --- lst encoder MLP (flat M = B*N) ---
  dim3 gflat((B * N) / 16, D / 64, 1);
  k_gemm_h<<<gflat, wv32, 0, stream>>>(lsth, 32, 0, lw1h, 32, 0, lb1,
                                       nullptr, 0, 0, l1h, D, 0, 0, 32, 1.f, 1);
  k_gemm_h<<<gflat, wv32, 0, stream>>>(l1h, D, 0, lw2h, D, 0, lb2,
                                       nullptr, 0, 0, l2h, D, 0, 0, D, 1.f, 1);
  k_gemm_h<<<gflat, wv32, 0, stream>>>(l2h, D, 0, lw3h, D, 0, lb3,
                                       lstA, D, 0, nullptr, 0, 0, 0, D, 1.f, 0);
  // --- supT = (features @ w_gcn)^T, f16 (B,D,N) ---
  dim3 gx(N / 16, D / 64, B);
  k_gemm_h<<<gx, wv32, 0, stream>>>(feath, 32, (long long)N * 32, wgcnh, 32, 0, nullptr,
                                    nullptr, 0, 0, supT, N, ND, 1, 32, 1.f, 0);
  // --- x0 = lrelu(g0 @ sup) ---
  k_gemm_h<<<gx, wv32, 0, stream>>>(g0, N, NN, supT, N, ND, nullptr,
                                    xbuf, D, ND, nullptr, 0, 0, 0, N, 1.f, 1);
  k_packx<<<(unsigned)(((long long)B * NP + 255) / 256), 256, 0, stream>>>(xbuf, xph, N, (long long)B * NP);

  const float scale = 1.f / sqrtf((float)D);
  float* lstCur = lstA;
  float* lstNxt = lstB;
  dim3 gs(N / 16, N / 64, B);

  for (int i = 0; i < L; ++i) {
    k_conv_h<<<gx, wv32, 0, stream>>>(xph, cwh[i][0], bq + i * D, nullptr, qh, 0, N);
    k_conv_h<<<gx, wv32, 0, stream>>>(xph, cwh[i][1], bk + i * D, nullptr, kh, 0, N);
    k_conv_h<<<gx, wv32, 0, stream>>>(xph, cwh[i][2], bv + i * D, lstCur, vth, 1, N);
    // scores = scale * q @ k^T   (kh row-major (m,d) == col-major B)
    k_gemm_h<<<gs, wv32, 0, stream>>>(qh, D, ND, kh, D, ND, nullptr,
                                      scores, N, NN, nullptr, 0, 0, 0, D, scale, 0);
    k_softmax<<<B * N, 256, 0, stream>>>(scores, dis, par, attn, N);
    // o = attn @ v  (vth is (B,D,N) col-major B)
    k_gemm_h<<<gx, wv32, 0, stream>>>(attn, N, NN, vth, N, ND, nullptr,
                                      lstNxt, D, ND, nullptr, 0, 0, 0, N, 1.f, 0);
    { float* tmp = lstNxt; lstNxt = lstCur; lstCur = tmp; }
    // x = LN(x + o)
    k_add_ln<<<B * N, 32, 0, stream>>>(xbuf, lstCur, g1 + i * D, b1 + i * D, xbuf, D);
    k_packx<<<(unsigned)(((long long)B * NP + 255) / 256), 256, 0, stream>>>(xbuf, xph, N, (long long)B * NP);
    // MLP (A = interior rows of xph)
    k_gemm_h<<<gx, wv32, 0, stream>>>(xph + 8 * D, D, NP, mwh[i][0], D, 0, mb1 + i * D,
                                      nullptr, 0, 0, m1h, D, ND, 0, D, 1.f, 1);
    k_gemm_h<<<gx, wv32, 0, stream>>>(m1h, D, ND, mwh[i][1], D, 0, mb2 + i * D,
                                      nullptr, 0, 0, m2h, D, ND, 0, D, 1.f, 1);
    k_gemm_h<<<gx, wv32, 0, stream>>>(m2h, D, ND, mwh[i][2], D, 0, mb3 + i * D,
                                      t0, D, ND, nullptr, 0, 0, 0, D, 1.f, 0);
    // x = LN(x + m)
    k_add_ln<<<B * N, 32, 0, stream>>>(xbuf, t0, g2 + i * D, b2 + i * D, xbuf, D);
    k_packx<<<(unsigned)(((long long)B * NP + 255) / 256), 256, 0, stream>>>(xbuf, xph, N, (long long)B * NP);
  }

  // --- output head ---
  k_gemm_h<<<gx, wv32, 0, stream>>>(xph + 8 * D, D, NP, ow1h, D, 0, ob1,
                                    t0, D, ND, nullptr, 0, 0, 0, D, 1.f, 0);
  k_add_ln<<<B * N, 32, 0, stream>>>(t0, nullptr, og, ob, t0, D);
  k_pack<<<(unsigned)(((long long)B * ND + 255) / 256), 256, 0, stream>>>(t0, th0, (long long)B * ND);
  k_gemm_h<<<gflat, wv32, 0, stream>>>(th0, D, 0, ow2h, D, 0, ob2,
                                       (float*)d_out, 256, 0, nullptr, 0, 0, 0, D, 1.f, 0);
}
